// MultiHeadSelfAttention_55911884259980
// MI455X (gfx1250) — compile-verified, matrix operand order checked by reading, not run on hardware
//
#include <hip/hip_runtime.h>
#include <hip/hip_bf16.h>

// ---------------------------------------------------------------------------
// Multi-head self-attention for MI455X (gfx1250).
// bf16 WMMA everywhere. GEMM tiles staged by the Tensor Data Mover
// (tensor_load_to_lds + TENSORcnt); attention K/V tiles staged by
// GLOBAL_LOAD_ASYNC_TO_LDS_B128 (ASYNCcnt). Fragments served from LDS.
// ---------------------------------------------------------------------------

typedef __bf16 bf16_t;
typedef __attribute__((ext_vector_type(8)))  __bf16 v8bf;
typedef __attribute__((ext_vector_type(16))) __bf16 v16bf;
typedef __attribute__((ext_vector_type(8)))  float  v8f;
typedef __attribute__((ext_vector_type(4)))  unsigned u32x4;
typedef __attribute__((ext_vector_type(8)))  unsigned u32x8;

static constexpr int Bc = 4;
static constexpr int Sc = 2048;
static constexpr int Dc = 1024;
static constexpr int Hc = 16;
static constexpr int HDc = 64;
static constexpr int BSc = Bc * Sc;

#define WMMA_BF16(A, Bm, C) \
  __builtin_amdgcn_wmma_f32_16x16x32_bf16(false, (A), false, (Bm), (short)0, (C), false, false)

#define AS3 __attribute__((address_space(3)))

__device__ __forceinline__ unsigned lds_addr_of(const void* p) {
  return (unsigned)(size_t)(AS3 const char*)p;  // LDS byte address
}

// One per-lane async 16-byte copy global -> LDS (tracked by ASYNCcnt).
__device__ __forceinline__ void async_copy_b128(const bf16_t* gsrc, bf16_t* ldst) {
  unsigned lds_off = lds_addr_of(ldst);
  asm volatile("global_load_async_to_lds_b128 %0, %1, off"
               :: "v"(lds_off), "v"(gsrc)
               : "memory");
}

// TDM: DMA a 2D tile (tile_rows x tile_cols bf16 elements, global row stride
// ld_elems) into LDS, packed row-major. D# per ISA cdna5 §8.3/8.4:
//  group0: count=1 | lds_addr | global_addr[56:0] | type=2
//  group1: data_size=2B; tensor dims huge (no OOB clipping); tile dims; stride
//  groups 2/3: zero (<=2D tensor, higher dims unused).
// Issued by the calling wave only (EXEC ignored); tracked by TENSORcnt.
__device__ __forceinline__ void tdm_load_2d(const bf16_t* gsrc, const bf16_t* ldst,
                                            int ld_elems, int tile_rows,
                                            int tile_cols) {
  const unsigned long long ga = (unsigned long long)(const void*)gsrc;
  u32x4 g0;
  g0[0] = 1u;                                           // count=1, user desc
  g0[1] = lds_addr_of(ldst);                            // lds_addr
  g0[2] = (unsigned)ga;                                 // global_addr[31:0]
  g0[3] = (unsigned)((ga >> 32) & 0x01FFFFFFu) | 0x80000000u;  // [56:32]|type=2
  u32x8 g1;
  g1[0] = 1u << 16;                     // data_size=1 (2 bytes), no multicast
  g1[1] = 0xFFFF0000u;                  // tensor_dim0 = 0x7FFFFFFF (lo16 here)
  g1[2] = 0xFFFF7FFFu;                  // tensor_dim0 hi / tensor_dim1 lo
  g1[3] = 0x7FFFu | ((unsigned)tile_cols << 16);  // tensor_dim1 hi | tile_dim0
  g1[4] = (unsigned)tile_rows;          // tile_dim1 | tile_dim2=0
  g1[5] = (unsigned)ld_elems;           // tensor_dim0_stride[31:0]
  g1[6] = 0u;                           // stride hi | tensor_dim1_stride lo
  g1[7] = 0u;
  u32x4 gz;
  gz[0] = 0u; gz[1] = 0u; gz[2] = 0u; gz[3] = 0u;
  asm volatile("tensor_load_to_lds %0, %1, %2, %3"
               :: "s"(g0), "s"(g1), "s"(gz), "s"(gz)
               : "memory");
}

// Load a 16x32 bf16 fragment (A-layout; same pattern serves B-layout since all
// our operands are stored K-contiguous). Per ISA 7.12.2: lanes 0-15 take K
// chunks {0..7,16..23}, lanes 16-31 take {8..15,24..31}; row = lane & 15.
__device__ __forceinline__ v16bf load_frag(const bf16_t* base, int row0, int k0,
                                           int ld, int lane) {
  const int r = lane & 15;
  const int h = (lane >> 4) & 1;
  const bf16_t* p = base + (size_t)(row0 + r) * ld + k0 + h * 8;
  v8bf lo = *reinterpret_cast<const v8bf*>(p);
  v8bf hi = *reinterpret_cast<const v8bf*>(p + 16);
  v16bf out;
#pragma unroll
  for (int i = 0; i < 8; ++i) { out[i] = lo[i]; out[i + 8] = hi[i]; }
  return out;
}

// ---------------------------------------------------------------------------
// fp32 -> bf16 conversion (vector-4 per thread; all sizes divisible by 4).
// ---------------------------------------------------------------------------
__global__ __launch_bounds__(256) void convert_bf16_kernel(
    const float* __restrict__ src, bf16_t* __restrict__ dst, int n) {
  int i = (blockIdx.x * 256 + threadIdx.x) * 4;
  if (i + 3 < n) {
    float4 f = *reinterpret_cast<const float4*>(src + i);
    dst[i + 0] = (bf16_t)f.x;
    dst[i + 1] = (bf16_t)f.y;
    dst[i + 2] = (bf16_t)f.z;
    dst[i + 3] = (bf16_t)f.w;
  }
}

// ---------------------------------------------------------------------------
// Tiled WMMA GEMM:  Y[m,n] = sum_k A[m,k] * W[n,k] + bias[n]
//   A:[M,K] bf16 row-major, W:[N,K] bf16 row-major (i.e. X @ W^T).
// Block = 256 threads (8 waves), tile 128(M) x 128(N); each wave does 32x64.
// K-step = 64: A/B 128x64 tiles DMA'd by the TDM (wave 0 issues
// tensor_load_to_lds, double-buffered on TENSORcnt). 16 WMMAs per barrier.
// mode 0: bf16 out, head-split [b,h,s,hd]
// mode 1: bf16 out, head-split transposed [b,h,hd,s]   (for V)
// mode 2: fp32 out, flat row-major [M,N]               (final projection)
// ---------------------------------------------------------------------------
__global__ __launch_bounds__(256) void wmma_gemm_kernel(
    const bf16_t* __restrict__ A, const bf16_t* __restrict__ W,
    const float* __restrict__ bias, void* __restrict__ out,
    int M, int N, int K, int mode) {
  __shared__ bf16_t sA[2][128 * 64];  // 16 KB per buffer
  __shared__ bf16_t sB[2][128 * 64];  // 16 KB per buffer

  const int tid = threadIdx.x;
  const int lane = tid & 31;
  const int wave = tid >> 5;
  const int mBlk = blockIdx.x * 128;
  const int nBlk = blockIdx.y * 128;
  const int mw = (wave >> 1) * 32;  // wave row offset inside block tile
  const int nw = (wave & 1) * 64;   // wave col offset inside block tile

  const int nk = K / 64;
  if (wave == 0) {  // prologue DMA: fill buffer 0
    tdm_load_2d(A + (size_t)mBlk * K, &sA[0][0], K, 128, 64);
    tdm_load_2d(W + (size_t)nBlk * K, &sB[0][0], K, 128, 64);
  }

  v8f c[2][4] = {};
  for (int kt = 0; kt < nk; ++kt) {
    const int buf = kt & 1;
    // Branch-free double buffering: always issue 2 TDM ops for the next
    // K-step (clamped on the last iteration -> harmless redundant refill),
    // then wait TENSORcnt<=2: the 2 older ops (this buffer) are complete.
    if (wave == 0) {
      const int knx = (kt + 1 < nk ? kt + 1 : kt) * 64;
      tdm_load_2d(A + (size_t)mBlk * K + knx, &sA[buf ^ 1][0], K, 128, 64);
      tdm_load_2d(W + (size_t)nBlk * K + knx, &sB[buf ^ 1][0], K, 128, 64);
      __builtin_amdgcn_s_wait_tensorcnt(2);
    }
    __syncthreads();

#pragma unroll
    for (int kk = 0; kk < 64; kk += 32) {
      v16bf a0 = load_frag(&sA[buf][0], mw, kk, 64, lane);
      v16bf a1 = load_frag(&sA[buf][0], mw + 16, kk, 64, lane);
      v16bf b[4];
#pragma unroll
      for (int j = 0; j < 4; ++j)
        b[j] = load_frag(&sB[buf][0], nw + j * 16, kk, 64, lane);
#pragma unroll
      for (int j = 0; j < 4; ++j) {
        c[0][j] = WMMA_BF16(a0, b[j], c[0][j]);
        c[1][j] = WMMA_BF16(a1, b[j], c[1][j]);
      }
    }
    __syncthreads();  // protect LDS buffer before next refill
  }

  // C/D layout (32-bit 16x16): lanes 0-15 -> rows 0..7, lanes 16-31 -> 8..15.
  const int h = (lane >> 4) & 1;
  const int col = lane & 15;
  const int m0 = mBlk + mw;
  const int n0 = nBlk + nw;
#pragma unroll
  for (int j = 0; j < 4; ++j) {
    const int n = n0 + j * 16 + col;
    const float bn = bias[n];
#pragma unroll
    for (int i = 0; i < 2; ++i) {
#pragma unroll
      for (int r = 0; r < 8; ++r) {
        const int m = m0 + i * 16 + r + h * 8;
        const float v = c[i][j][r] + bn;
        if (mode == 2) {
          reinterpret_cast<float*>(out)[(size_t)m * N + n] = v;
        } else {
          const int bb = m / Sc, ss = m % Sc;
          const int hh = n / HDc, hd = n % HDc;
          const size_t idx =
              (mode == 0) ? ((size_t)(bb * Hc + hh) * Sc + ss) * HDc + hd
                          : ((size_t)(bb * Hc + hh) * HDc + hd) * Sc + ss;
          reinterpret_cast<bf16_t*>(out)[idx] = (bf16_t)v;
        }
      }
    }
  }
}

// ---------------------------------------------------------------------------
// Flash attention. Grid: (B*H, S/128). 8 waves/block, each wave owns 16 q-rows.
// Q:[b,h,s,hd], K:[b,h,s,hd], Vt:[b,h,hd,s].
// K/V tiles (32 keys) async-staged (ASYNCcnt) into double-buffered LDS shared
// by all waves; waves 0-3 fill K, 4-7 fill V -> uniform 2 async ops per wave
// per step, branch-free clamped double buffering. 4 WMMAs for Q.K^T, fp32
// online softmax, P via wave-private LDS (C-layout -> A-frag), 4 WMMAs P.V.
// ---------------------------------------------------------------------------
__global__ __launch_bounds__(256) void flash_attn_kernel(
    const bf16_t* __restrict__ Q, const bf16_t* __restrict__ Kmat,
    const bf16_t* __restrict__ Vt, const int* __restrict__ mask,
    bf16_t* __restrict__ attn_out) {
  __shared__ bf16_t sK[2][32 * 64];     // 4 KB per buffer: 32 keys x 64 hd
  __shared__ bf16_t sV[2][64 * 32];     // 4 KB per buffer: 64 hd x 32 keys
  __shared__ bf16_t pTile[8][16 * 32];  // per-wave P tile

  const int tid = threadIdx.x;
  const int lane = tid & 31;
  const int wave = tid >> 5;
  const int bh = blockIdx.x;
  const int b = bh / Hc;
  const int hh = bh % Hc;
  const int q0 = blockIdx.y * 128 + wave * 16;

  const bf16_t* Qb = Q + ((size_t)bh * Sc + q0) * HDc;
  const bf16_t* Kb = Kmat + (size_t)bh * Sc * HDc;
  const bf16_t* Vb = Vt + (size_t)bh * HDc * Sc;
  const int* mb = mask + b * Sc;

  // Stage one 32-key step: waves 0-3 -> K tile, waves 4-7 -> V tile.
  auto stage = [&](int buf, int kb) {
    if (tid < 128) {
#pragma unroll
      for (int i = 0; i < 2; ++i) {
        const int id = tid + 128 * i;          // 0..255
        const int row = id >> 3, ch = id & 7;  // 32 rows x 8 chunks
        async_copy_b128(Kb + (size_t)(kb + row) * HDc + ch * 8,
                        &sK[buf][row * 64 + ch * 8]);
      }
    } else {
#pragma unroll
      for (int i = 0; i < 2; ++i) {
        const int id = (tid - 128) + 128 * i;  // 0..255
        const int row = id >> 2, ch = id & 3;  // 64 rows x 4 chunks
        async_copy_b128(Vb + (size_t)row * Sc + kb + ch * 8,
                        &sV[buf][row * 32 + ch * 8]);
      }
    }
  };

  // Q fragments for this wave's 16 rows (Hd = 64 -> two K=32 fragments).
  const v16bf aq0 = load_frag(Qb, 0, 0, HDc, lane);
  const v16bf aq1 = load_frag(Qb, 0, 32, HDc, lane);

  v8f o[4] = {};
  float mrow[8], lrow[8];
#pragma unroll
  for (int r = 0; r < 8; ++r) { mrow[r] = -3.0e38f; lrow[r] = 0.0f; }

  const int col = lane & 15;
  const int h = (lane >> 4) & 1;
  bf16_t* pw = &pTile[wave][0];

  stage(0, 0);
  const int nkb = Sc / 32;
  for (int kt = 0; kt < nkb; ++kt) {
    const int buf = kt & 1;
    const int kb = kt * 32;
    // Branch-free: always stage next (clamped), always wait <= 2.
    stage(buf ^ 1, (kt + 1 < nkb ? kt + 1 : kt) * 32);
    asm volatile("s_wait_asynccnt 0x2" ::: "memory");
    __syncthreads();

    // ---- scores: two 16x16 tiles over 32 keys (fragments from LDS) -----
    v16bf k00 = load_frag(&sK[buf][0], 0, 0, 64, lane);
    v16bf k01 = load_frag(&sK[buf][0], 0, 32, 64, lane);
    v16bf k10 = load_frag(&sK[buf][0], 16, 0, 64, lane);
    v16bf k11 = load_frag(&sK[buf][0], 16, 32, 64, lane);
    v8f s0 = {}, s1 = {};
    s0 = WMMA_BF16(aq0, k00, s0);
    s0 = WMMA_BF16(aq1, k01, s0);
    s1 = WMMA_BF16(aq0, k10, s1);
    s1 = WMMA_BF16(aq1, k11, s1);

    // scale 1/sqrt(64) and key mask (column = lane within tile)
    const bool keep0 = mb[kb + col] != 0;
    const bool keep1 = mb[kb + 16 + col] != 0;
#pragma unroll
    for (int r = 0; r < 8; ++r) {
      s0[r] = keep0 ? s0[r] * 0.125f : -1.0e9f;
      s1[r] = keep1 ? s1[r] * 0.125f : -1.0e9f;
    }

    // ---- online softmax (row stats replicated across each 16-lane half) --
    float mnew[8], alpha[8];
#pragma unroll
    for (int r = 0; r < 8; ++r) {
      float v = fmaxf(s0[r], s1[r]);
      v = fmaxf(v, __shfl_xor(v, 1, 32));
      v = fmaxf(v, __shfl_xor(v, 2, 32));
      v = fmaxf(v, __shfl_xor(v, 4, 32));
      v = fmaxf(v, __shfl_xor(v, 8, 32));
      mnew[r] = fmaxf(mrow[r], v);
      alpha[r] = __expf(mrow[r] - mnew[r]);
      mrow[r] = mnew[r];
    }
    float p0[8], p1[8];
#pragma unroll
    for (int r = 0; r < 8; ++r) {
      p0[r] = __expf(s0[r] - mnew[r]);
      p1[r] = __expf(s1[r] - mnew[r]);
      float rs = p0[r] + p1[r];
      rs += __shfl_xor(rs, 1, 32);
      rs += __shfl_xor(rs, 2, 32);
      rs += __shfl_xor(rs, 4, 32);
      rs += __shfl_xor(rs, 8, 32);
      lrow[r] = lrow[r] * alpha[r] + rs;
    }
#pragma unroll
    for (int t = 0; t < 4; ++t)
#pragma unroll
      for (int r = 0; r < 8; ++r) o[t][r] *= alpha[r];

    // ---- re-layout P (C-layout -> A-fragment) via wave-private LDS ------
#pragma unroll
    for (int r = 0; r < 8; ++r) {
      const int row = r + h * 8;
      pw[row * 32 + col] = (bf16_t)p0[r];
      pw[row * 32 + 16 + col] = (bf16_t)p1[r];
    }
    // Same-wave LDS ops are in-order (ISA 7.3): reload safe w/o barrier.
    v16bf pa = load_frag(pw, 0, 0, 32, lane);

    // ---- O += P @ V (V fragments from LDS, key-contiguous) --------------
#pragma unroll
    for (int t = 0; t < 4; ++t) {
      v16bf vf = load_frag(&sV[buf][0], t * 16, 0, 32, lane);
      o[t] = WMMA_BF16(pa, vf, o[t]);
    }
    __syncthreads();  // protect sK/sV before next refill
  }

  // ---- finalize: divide by l, write attn_out flat [B*S, D] (bf16) --------
#pragma unroll
  for (int r = 0; r < 8; ++r) lrow[r] = lrow[r] > 0.0f ? 1.0f / lrow[r] : 0.0f;
#pragma unroll
  for (int t = 0; t < 4; ++t) {
#pragma unroll
    for (int r = 0; r < 8; ++r) {
      const int qrow = q0 + r + h * 8;
      const size_t idx = ((size_t)b * Sc + qrow) * Dc + hh * HDc + t * 16 + col;
      attn_out[idx] = (bf16_t)(o[t][r] * lrow[r]);
    }
  }
}

// ---------------------------------------------------------------------------
// Host launcher.
// ---------------------------------------------------------------------------
extern "C" void kernel_launch(void* const* d_in, const int* in_sizes, int n_in,
                              void* d_out, int out_size, void* d_ws,
                              size_t ws_size, hipStream_t stream) {
  const float* x = (const float*)d_in[0];
  const int* mask = (const int*)d_in[1];
  const float* Wq = (const float*)d_in[2];
  const float* bq = (const float*)d_in[3];
  const float* Wk = (const float*)d_in[4];
  const float* bk = (const float*)d_in[5];
  const float* Wv = (const float*)d_in[6];
  const float* bv = (const float*)d_in[7];
  const float* Wo = (const float*)d_in[8];
  const float* bo = (const float*)d_in[9];
  float* out = (float*)d_out;

  // Workspace carve-up (bf16): x(16M) + 4 weights(8M) + Q,K,Vt(48M) + attn(16M)
  char* ws = (char*)d_ws;
  bf16_t* xb = (bf16_t*)ws;   ws += (size_t)BSc * Dc * sizeof(bf16_t);
  bf16_t* Wqb = (bf16_t*)ws;  ws += (size_t)Dc * Dc * sizeof(bf16_t);
  bf16_t* Wkb = (bf16_t*)ws;  ws += (size_t)Dc * Dc * sizeof(bf16_t);
  bf16_t* Wvb = (bf16_t*)ws;  ws += (size_t)Dc * Dc * sizeof(bf16_t);
  bf16_t* Wob = (bf16_t*)ws;  ws += (size_t)Dc * Dc * sizeof(bf16_t);
  bf16_t* Qb = (bf16_t*)ws;   ws += (size_t)BSc * Dc * sizeof(bf16_t);
  bf16_t* Kb = (bf16_t*)ws;   ws += (size_t)BSc * Dc * sizeof(bf16_t);
  bf16_t* Vtb = (bf16_t*)ws;  ws += (size_t)BSc * Dc * sizeof(bf16_t);
  bf16_t* Attnb = (bf16_t*)ws;

  // 1) fp32 -> bf16 conversions
  {
    int n = BSc * Dc;
    convert_bf16_kernel<<<n / 1024, 256, 0, stream>>>(x, xb, n);
    n = Dc * Dc;
    convert_bf16_kernel<<<n / 1024, 256, 0, stream>>>(Wq, Wqb, n);
    convert_bf16_kernel<<<n / 1024, 256, 0, stream>>>(Wk, Wkb, n);
    convert_bf16_kernel<<<n / 1024, 256, 0, stream>>>(Wv, Wvb, n);
    convert_bf16_kernel<<<n / 1024, 256, 0, stream>>>(Wo, Wob, n);
  }

  // 2) QKV projections (WMMA GEMM, TDM-staged)
  dim3 gg(BSc / 128, Dc / 128);
  wmma_gemm_kernel<<<gg, 256, 0, stream>>>(xb, Wqb, bq, Qb, BSc, Dc, Dc, 0);
  wmma_gemm_kernel<<<gg, 256, 0, stream>>>(xb, Wkb, bk, Kb, BSc, Dc, Dc, 0);
  wmma_gemm_kernel<<<gg, 256, 0, stream>>>(xb, Wvb, bv, Vtb, BSc, Dc, Dc, 1);

  // 3) Flash attention (async-LDS staged)
  flash_attn_kernel<<<dim3(Bc * Hc, Sc / 128), 256, 0, stream>>>(
      Qb, Kb, Vtb, mask, Attnb);

  // 4) Output projection (fp32 out + bias)
  wmma_gemm_kernel<<<gg, 256, 0, stream>>>(Attnb, Wob, bo, out, BSc, Dc, Dc, 2);
}
